// FastRCNN_73418170958447
// MI455X (gfx1250) — compile-verified, matrix operand order checked by reading, not run on hardware
//
#include <hip/hip_runtime.h>
#include <hip/hip_bf16.h>
#include <cstdint>
#include <cstddef>

typedef __attribute__((ext_vector_type(16))) _Float16 v16h;
typedef __attribute__((ext_vector_type(8)))  _Float16 v8h;
typedef __attribute__((ext_vector_type(8)))  float    v8f;

#define N_ANCH   25
#define NROI     625        // 25*5*5
#define BATCH    2
#define M_ROWS   1250       // BATCH * NROI
#define M_PAD    1280       // 40 macro-tiles of 32
#define CCH      512
#define FEATW    7
#define KDIM     25088      // 512*49, = 784 * 32
#define NOUT     200
#define N_PAD    224        // 7 macro-tiles of 32
#define KCLS     17

// ---------------------------------------------------------------------------
// Kernel 1: anchor decode + conv/MP coordinate tracking -> per-ROI bin bounds
// ---------------------------------------------------------------------------
__device__ __forceinline__ int fdiv7(int a) {
    return (a >= 0) ? (a / 7) : -(((-a) + 6) / 7);   // python floor-div by 7
}

__global__ void k_decode(const float* __restrict__ regcls,
                         const float* __restrict__ wA, const float* __restrict__ hA,
                         const float* __restrict__ xA, const float* __restrict__ yA,
                         int* __restrict__ meta) {
    int m = blockIdx.x * blockDim.x + threadIdx.x;
    if (m >= M_ROWS) return;
    int b = m / NROI, n = m % NROI;
    const float* rc = regcls + (size_t)b * 5 * NROI;
    float r0 = rc[0 * NROI + n], r1 = rc[1 * NROI + n];
    float r2 = rc[2 * NROI + n], r3 = rc[3 * NROI + n];
    float r4 = rc[4 * NROI + n];
    float wa = wA[n], ha = hA[n], xa = xA[n], ya = yA[n];

    float wreg = expf(r2) * wa;
    float hreg = expf(r3) * ha;
    float xreg = r0 * wa + xa;
    float yreg = r1 * ha + ya;
    float xis = xreg - 0.5f * wreg, xfs = xreg + 0.5f * wreg;
    float yis = yreg - 0.5f * hreg, yfs = yreg + 0.5f * hreg;

    float c0 = floorf(fmaxf(xis, 0.f));
    float c1 = floorf(fmaxf(yis, 0.f));
    float c2 = fminf(ceilf(xfs), 295.f);
    float c3 = fmaxf(ceilf(yfs), 295.f);       // (sic) max, per reference
    bool valid = (xis < 296.f) && (yis < 296.f) && (xfs >= 0.f) && (yfs >= 0.f);
    float sg = 1.f / (1.f + expf(-r4));
    int on = (valid && sg > 0.f) ? 1 : 0;

    // sizes after each op of ['c','c','M','c','c','M','c','c','c','M','c','c','c','M','c','c','c']
    const int   opv[17] = {0,0,1,0,0,1,0,0,0,1,0,0,0,1,0,0,0};
    const float szm[17] = {293.f,291.f,145.f,143.f,141.f,70.f,68.f,66.f,64.f,
                           31.f,29.f,27.f,25.f,12.f,10.f,8.f,6.f}; // size-1
#pragma unroll
    for (int i = 0; i < 17; ++i) {
        if (opv[i] == 0) {
            c0 = fmaxf(c0 - 1.f, 0.f);
            c1 = fmaxf(c1 - 1.f, 0.f);
            c2 = fminf(c2, szm[i]);
            c3 = fminf(c3, szm[i]);
        } else {
            c0 = floorf(c0 * 0.5f); c1 = floorf(c1 * 0.5f);
            c2 = floorf(c2 * 0.5f); c3 = floorf(c3 * 0.5f);
        }
    }
    int xi = (int)c0, yi = (int)c1, xf = (int)c2, yf = (int)c3;
    int wx = xf + 1 - xi, wy = yf + 1 - yi;
    int* mm = meta + (size_t)m * 18;
#pragma unroll
    for (int ii = 0; ii < 8; ++ii) {
        mm[ii]     = xi + fdiv7(ii * wx);
        mm[8 + ii] = yi + fdiv7(ii * wy);
    }
    mm[16] = on;
}

// ---------------------------------------------------------------------------
// Kernel 2: 7x7 RoI max-pool -> fp16 A matrix [M_PAD x KDIM], k = c*49+i*7+j
// ---------------------------------------------------------------------------
__global__ __launch_bounds__(256)
void k_pool(const float* __restrict__ cfeats, const int* __restrict__ meta,
            _Float16* __restrict__ poolA) {
    __shared__ int sbx[8], sby[8], son;
    int m = blockIdx.x;
    if (threadIdx.x == 0) {
        const int* mm = meta + (size_t)m * 18;
        for (int i = 0; i < 8; ++i) { sbx[i] = mm[i]; sby[i] = mm[8 + i]; }
        son = mm[16];
    }
    __syncthreads();
    int b = m / NROI;
    const float* F = cfeats + (size_t)b * CCH * 49;
    for (int c = threadIdx.x; c < CCH; c += blockDim.x) {
        const float* fc = F + (size_t)c * 49;
        float v1[7][7];   // [bin_i][y] : max over x within bin i
#pragma unroll
        for (int i = 0; i < 7; ++i)
#pragma unroll
            for (int y = 0; y < 7; ++y) v1[i][y] = -1e30f;
#pragma unroll
        for (int x = 0; x < 7; ++x) {
            float row[7];
#pragma unroll
            for (int y = 0; y < 7; ++y) row[y] = fc[x * 7 + y];
#pragma unroll
            for (int i = 0; i < 7; ++i) {
                if (x >= sbx[i] && x < sbx[i + 1]) {
#pragma unroll
                    for (int y = 0; y < 7; ++y) v1[i][y] = fmaxf(v1[i][y], row[y]);
                }
            }
        }
        _Float16* out = poolA + (size_t)m * KDIM + (size_t)c * 49;
#pragma unroll
        for (int i = 0; i < 7; ++i) {
#pragma unroll
            for (int j = 0; j < 7; ++j) {
                float v = -1e30f;
#pragma unroll
                for (int y = 0; y < 7; ++y)
                    if (y >= sby[j] && y < sby[j + 1]) v = fmaxf(v, v1[i][y]);
                float pv = (son && v > -5e29f) ? v : 0.f;
                out[i * 7 + j] = (_Float16)pv;
            }
        }
    }
}

// ---------------------------------------------------------------------------
// Kernel 3: fp32 -> fp16 weight conversion for w1
// ---------------------------------------------------------------------------
__global__ void k_h16(const float* __restrict__ src, _Float16* __restrict__ dst, int n) {
    int i = blockIdx.x * blockDim.x + threadIdx.x;
    if (i < n) dst[i] = (_Float16)src[i];
}

// ---------------------------------------------------------------------------
// Kernel 4: WMMA GEMM  z1buf[M_PAD x N_PAD] = A[M_PAD x K] * W1h[N_PAD x K]^T
// 4 waves / block; each wave owns a 32x32 macro-tile = 2x2 register tiles.
// Per K-step of 32: 8 x 16B loads feed 4 independent v_wmma chains
// (2 loads per wmma, 4-way ILP to hide XDL accumulate latency).
// ---------------------------------------------------------------------------
__global__ __launch_bounds__(128)
void k_gemm(const _Float16* __restrict__ A, const _Float16* __restrict__ Bm,
            float* __restrict__ Cm) {
    const int wave  = threadIdx.x >> 5;
    const int lane  = threadIdx.x & 31;
    const int mbase = (blockIdx.x * 4 + wave) * 32;   // 2 m-tiles of 16
    const int nbase = blockIdx.y * 32;                // 2 n-tiles of 16

    // 16-bit A/B fragment K-mapping: lane<16 -> K{0..7,16..23}; lane>=16 -> +8
    const int k0  = (lane & 16) ? 8 : 0;
    const int sub = lane & 15;
    const _Float16* pa0 = A  + (size_t)(mbase + sub) * KDIM + k0;
    const _Float16* pa1 = pa0 + (size_t)16 * KDIM;
    const _Float16* pb0 = Bm + (size_t)(nbase + sub) * KDIM + k0;
    const _Float16* pb1 = pb0 + (size_t)16 * KDIM;

    v8f acc00 = {}, acc01 = {}, acc10 = {}, acc11 = {};
    for (int k = 0; k < KDIM; k += 32) {
        union { v16h v; v8h h[2]; } a0, a1, b0, b1;
        a0.h[0] = *(const v8h*)(pa0 + k);  a0.h[1] = *(const v8h*)(pa0 + k + 16);
        a1.h[0] = *(const v8h*)(pa1 + k);  a1.h[1] = *(const v8h*)(pa1 + k + 16);
        b0.h[0] = *(const v8h*)(pb0 + k);  b0.h[1] = *(const v8h*)(pb0 + k + 16);
        b1.h[0] = *(const v8h*)(pb1 + k);  b1.h[1] = *(const v8h*)(pb1 + k + 16);
        __builtin_prefetch(pa0 + k + 256, 0, 3);   // global_prefetch_b8
        __builtin_prefetch(pa1 + k + 256, 0, 3);
        __builtin_prefetch(pb0 + k + 256, 0, 3);
        __builtin_prefetch(pb1 + k + 256, 0, 3);
        acc00 = __builtin_amdgcn_wmma_f32_16x16x32_f16(
                    false, a0.v, false, b0.v, (short)0, acc00, false, false);
        acc01 = __builtin_amdgcn_wmma_f32_16x16x32_f16(
                    false, a0.v, false, b1.v, (short)0, acc01, false, false);
        acc10 = __builtin_amdgcn_wmma_f32_16x16x32_f16(
                    false, a1.v, false, b0.v, (short)0, acc10, false, false);
        acc11 = __builtin_amdgcn_wmma_f32_16x16x32_f16(
                    false, a1.v, false, b1.v, (short)0, acc11, false, false);
    }

    // D layout: VGPR r holds M = r + 8*(lane>=16), N = lane&15
    const int col   = lane & 15;
    const int rbase = (lane >> 4) * 8;
    float* o00 = Cm + (size_t)(mbase +  0) * N_PAD + nbase +  0 + col;
    float* o01 = Cm + (size_t)(mbase +  0) * N_PAD + nbase + 16 + col;
    float* o10 = Cm + (size_t)(mbase + 16) * N_PAD + nbase +  0 + col;
    float* o11 = Cm + (size_t)(mbase + 16) * N_PAD + nbase + 16 + col;
#pragma unroll
    for (int r = 0; r < 8; ++r) {
        size_t roff = (size_t)(rbase + r) * N_PAD;
        o00[roff] = acc00[r];
        o01[roff] = acc01[r];
        o10[roff] = acc10[r];
        o11[roff] = acc11[r];
    }
}

// ---------------------------------------------------------------------------
// Kernel 5: MLP tail per ROI: relu(z1+b1) -> relu(@w2+b2) -> heads w3/w4
// ---------------------------------------------------------------------------
__global__ __launch_bounds__(128)
void k_head(const float* __restrict__ z1buf,
            const float* __restrict__ b1, const float* __restrict__ w2,
            const float* __restrict__ b2, const float* __restrict__ w3,
            const float* __restrict__ b3, const float* __restrict__ w4,
            const float* __restrict__ b4, float* __restrict__ outp) {
    __shared__ float sz[NOUT];
    __shared__ float s2[100];
    const int m = blockIdx.x;
    const int t = threadIdx.x;
    const float* zr = z1buf + (size_t)m * N_PAD;
    for (int i = t; i < NOUT; i += blockDim.x)
        sz[i] = fmaxf(zr[i] + b1[i], 0.f);
    __syncthreads();
    for (int o = t; o < 100; o += blockDim.x) {
        const float* wr = w2 + (size_t)o * 200;
        float s = b2[o];
        for (int k = 0; k < 200; ++k) s = fmaf(sz[k], wr[k], s);
        s2[o] = fmaxf(s, 0.f);
    }
    __syncthreads();
    const int b = m / NROI, n = m % NROI;
    for (int o = t; o < 85; o += blockDim.x) {
        if (o < KCLS) {
            const float* wr = w3 + (size_t)o * 100;
            float s = b3[o];
            for (int k = 0; k < 100; ++k) s = fmaf(s2[k], wr[k], s);
            outp[(size_t)m * KCLS + o] = s;
        } else {
            int o2 = o - KCLS;
            const float* wr = w4 + (size_t)o2 * 100;
            float s = b4[o2];
            for (int k = 0; k < 100; ++k) s = fmaf(s2[k], wr[k], s);
            int sg = o2 / KCLS, kk = o2 % KCLS;
            outp[(size_t)M_ROWS * KCLS +
                 (((size_t)(b * 4 + sg)) * NROI + n) * KCLS + kk] = s;
        }
    }
}

// ---------------------------------------------------------------------------
extern "C" void kernel_launch(void* const* d_in, const int* in_sizes, int n_in,
                              void* d_out, int out_size, void* d_ws, size_t ws_size,
                              hipStream_t stream) {
    const float* cfeats = (const float*)d_in[0];
    const float* regcls = (const float*)d_in[1];
    const float* wanch  = (const float*)d_in[2];
    const float* hanch  = (const float*)d_in[3];
    const float* xanch  = (const float*)d_in[4];
    const float* yanch  = (const float*)d_in[5];
    const float* w1 = (const float*)d_in[6];
    const float* b1 = (const float*)d_in[7];
    const float* w2 = (const float*)d_in[8];
    const float* b2 = (const float*)d_in[9];
    const float* w3 = (const float*)d_in[10];
    const float* b3 = (const float*)d_in[11];
    const float* w4 = (const float*)d_in[12];
    const float* b4 = (const float*)d_in[13];
    float* out = (float*)d_out;
    (void)in_sizes; (void)n_in; (void)out_size; (void)ws_size;

    // workspace carve-up (all 256-B aligned)
    char* ws = (char*)d_ws;
    int*      meta  = (int*)ws;                                  // 1250*18*4    = 90000 B
    _Float16* poolA = (_Float16*)(ws + 90112);                   // 1280*25088*2 = 64225280 B
    _Float16* w1h   = (_Float16*)(ws + 90112 + 64225280);        // 224*25088*2  = 11239424 B
    float*    z1buf = (float*)   (ws + 90112 + 64225280 + 11239424); // 1280*224*4 = 1146880 B

    // 1. decode ROIs
    k_decode<<<(M_ROWS + 127) / 128, 128, 0, stream>>>(regcls, wanch, hanch,
                                                       xanch, yanch, meta);
    // 2. RoI pool -> fp16 A matrix
    k_pool<<<M_ROWS, 256, 0, stream>>>(cfeats, meta, poolA);
    // 3. w1 -> fp16
    {
        int n = NOUT * KDIM;
        k_h16<<<(n + 255) / 256, 256, 0, stream>>>(w1, w1h, n);
    }
    // 4. big GEMM via WMMA: 40 M-macro-tiles (4 waves/block) x 7 N-macro-tiles
    dim3 gg(M_PAD / 32 / 4, N_PAD / 32);
    k_gemm<<<gg, 128, 0, stream>>>(poolA, w1h, z1buf);
    // 5. MLP tail + output scatter
    k_head<<<M_ROWS, 128, 0, stream>>>(z1buf, b1, w2, b2, w3, b3, w4, b4, out);
}
